// ConformerFFNMetal_3856880632425
// MI455X (gfx1250) — matrix operand order repaired and checked by reading, not compiled, not run on hardware
//
#include <hip/hip_runtime.h>
#include <hip/hip_bf16.h>
#include <math.h>

// ---------------------------------------------------------------------------
// ConformerFFN:  LN -> (int8-dequant) FC1 -> SiLU -> (int8-dequant) FC2
//                -> residual + 0.5*z
// B=8 S=2048 H=1024 F=4096 G=128, M = B*S = 16384
// ---------------------------------------------------------------------------

typedef __attribute__((ext_vector_type(16))) _Float16 v16h;
typedef __attribute__((ext_vector_type(8)))  _Float16 h8;
typedef __attribute__((ext_vector_type(4)))  _Float16 h4;
typedef __attribute__((ext_vector_type(8)))  float    v8f;

#define EPS 1e-5f

// ---- CDNA5 async global->LDS copy (ASYNCcnt-tracked), with fallback -------
#if defined(__has_builtin)
#if __has_builtin(__builtin_amdgcn_global_load_async_to_lds_b128)
#define HAVE_ASYNC_LDS 1
#endif
#endif
#ifndef HAVE_ASYNC_LDS
#define HAVE_ASYNC_LDS 0
#endif

#if HAVE_ASYNC_LDS
typedef int v4i_g __attribute__((vector_size(16)));
__device__ __forceinline__ void async_copy_b128(const _Float16* g, _Float16* l)
{
    __builtin_amdgcn_global_load_async_to_lds_b128(
        (__attribute__((address_space(1))) v4i_g*)g,
        (__attribute__((address_space(3))) v4i_g*)l,
        0, 0);
}
__device__ __forceinline__ void wait_async0()
{
#if __has_builtin(__builtin_amdgcn_s_wait_asynccnt)
    __builtin_amdgcn_s_wait_asynccnt(0);
#else
    asm volatile("s_wait_asynccnt 0x0" ::: "memory");
#endif
}
#endif

// ---------------------------------------------------------------------------
// Kernel 1: LayerNorm (fp32 in) -> f16 activations
// ---------------------------------------------------------------------------
__global__ __launch_bounds__(256)
void ffn_ln_kernel(const float* __restrict__ x,
                   const float* __restrict__ gamma,
                   const float* __restrict__ beta,
                   _Float16* __restrict__ h_out)
{
    const int row = blockIdx.x;
    const int t   = threadIdx.x;
    const size_t base = (size_t)row * 1024 + t * 4;

    float4 v = *(const float4*)(x + base);
    float s  = v.x + v.y + v.z + v.w;
    float sq = v.x * v.x + v.y * v.y + v.z * v.z + v.w * v.w;

    __shared__ float2 red[256];
    red[t] = make_float2(s, sq);
    __syncthreads();
    #pragma unroll
    for (int st = 128; st > 0; st >>= 1) {
        if (t < st) {
            float2 a = red[t], b = red[t + st];
            red[t] = make_float2(a.x + b.x, a.y + b.y);
        }
        __syncthreads();
    }
    const float mean = red[0].x * (1.0f / 1024.0f);
    const float var  = red[0].y * (1.0f / 1024.0f) - mean * mean;
    const float rstd = rsqrtf(var + EPS);

    float4 g  = *(const float4*)(gamma + t * 4);
    float4 bb = *(const float4*)(beta + t * 4);

    h4 o;
    o[0] = (_Float16)((v.x - mean) * rstd * g.x + bb.x);
    o[1] = (_Float16)((v.y - mean) * rstd * g.y + bb.y);
    o[2] = (_Float16)((v.z - mean) * rstd * g.z + bb.z);
    o[3] = (_Float16)((v.w - mean) * rstd * g.w + bb.w);
    *(h4*)(h_out + base) = o;
}

// ---------------------------------------------------------------------------
// Kernel 2: dequantize + TRANSPOSE weights to f16.
//   in : q [K,N] int32 (zp=128), scales [K/128, N]
//   out: wt [N,K] f16
// ---------------------------------------------------------------------------
#define TDK 64
#define TDN 64
#define TLD 72   // padded LDS row stride (144B, keeps h8 reads 16B-aligned)

__global__ __launch_bounds__(256)
void ffn_dequant_t_kernel(const int* __restrict__ q,
                          const float* __restrict__ scales,
                          _Float16* __restrict__ wt,
                          int K, int N)
{
    __shared__ __align__(16) _Float16 tileT[TDN * TLD];   // tileT[n][k]

    const int tid = threadIdx.x;
    const int kb  = blockIdx.y * TDK;
    const int nb  = blockIdx.x * TDN;

    #pragma unroll
    for (int p = 0; p < 4; ++p) {
        const int v  = tid + p * 256;
        const int kr = v >> 4;            // 0..63
        const int n  = (v & 15) << 2;     // 0,4,...,60
        const int k  = kb + kr;
        const int g  = k >> 7;            // G = 128

        int4   qv = *(const int4*)(q + (size_t)k * N + nb + n);
        float4 sv = *(const float4*)(scales + (size_t)g * N + nb + n);

        tileT[(n + 0) * TLD + kr] = (_Float16)((float)(qv.x - 128) * sv.x);
        tileT[(n + 1) * TLD + kr] = (_Float16)((float)(qv.y - 128) * sv.y);
        tileT[(n + 2) * TLD + kr] = (_Float16)((float)(qv.z - 128) * sv.z);
        tileT[(n + 3) * TLD + kr] = (_Float16)((float)(qv.w - 128) * sv.w);
    }
    __syncthreads();

    #pragma unroll
    for (int p = 0; p < 2; ++p) {
        const int v  = tid + p * 256;
        const int nr = v >> 3;            // 0..63
        const int kc = (v & 7) << 3;      // 0,8,...,56
        *(h8*)(wt + (size_t)(nb + nr) * K + kb + kc) =
            *(const h8*)&tileT[nr * TLD + kc];
    }
}

// ---------------------------------------------------------------------------
// Kernel 3: double-buffered WMMA GEMM, C[M,N] = A[M,K](f16) * Bt[N,K](f16)
//   EPI==1 : out16 = f16( silu(acc + bias[n]) )          (FC1)
//   EPI==2 : out32 = residual + 0.5f*(acc + bias[n])     (FC2)
// Block = 256 threads = 8 waves; tile BM=128 BN=128 BK=32; staging via
// GLOBAL_LOAD_ASYNC_TO_LDS_B128 (ASYNCcnt) when available.
// ---------------------------------------------------------------------------
#define BM 128
#define BN 128
#define BK 32
#define LDT 40   // padded LDS row stride in halves (80B: conflict-free b128)

template <int EPI>
__global__ __launch_bounds__(256)
void ffn_gemm_kernel(const _Float16* __restrict__ A,
                     const _Float16* __restrict__ Bt,   // [N,K]
                     const float*    __restrict__ bias,
                     const float*    __restrict__ residual,
                     _Float16*       __restrict__ out16,
                     float*          __restrict__ out32,
                     int M, int N, int K)
{
    __shared__ __align__(16) _Float16 As[2][BM * LDT];
    __shared__ __align__(16) _Float16 Bs[2][BN * LDT];   // Bs[n][k]

    const int tid  = threadIdx.x;
    const int lane = tid & 31;
    const int wave = tid >> 5;
    const int wm   = wave & 3;           // 32-row band
    const int wn   = wave >> 2;          // 64-col band
    const int half = lane >> 4;          // 0 | 1
    const int mrow = lane & 15;
    const int cA   = half << 3;          // A frag K base: 0 | 8
    const int cB   = half << 4;          // unused (B frags read at cA: 0 | 8)

    const int bm = blockIdx.y * BM;
    const int bn = blockIdx.x * BN;

    // per-thread staging slice: 2 x h8 from A, 2 x h8 from Bt
    const int row0 = tid >> 2;                 // 0..63
    const int row1 = row0 + 64;                // 64..127
    const int kc   = (tid & 3) << 3;           // 0,8,16,24

    const _Float16* pa0 = A  + (size_t)(bm + row0) * K + kc;
    const _Float16* pa1 = A  + (size_t)(bm + row1) * K + kc;
    const _Float16* pb0 = Bt + (size_t)(bn + row0) * K + kc;
    const _Float16* pb1 = Bt + (size_t)(bn + row1) * K + kc;

    _Float16* sa0 = &As[0][row0 * LDT + kc];
    _Float16* sa1 = &As[0][row1 * LDT + kc];
    _Float16* sb0 = &Bs[0][row0 * LDT + kc];
    _Float16* sb1 = &Bs[0][row1 * LDT + kc];
    const int bufStride = BM * LDT;            // halves between buffers

    v8f acc[2][4];
    #pragma unroll
    for (int i = 0; i < 2; ++i)
        #pragma unroll
        for (int j = 0; j < 4; ++j) {
            v8f z = {0.f, 0.f, 0.f, 0.f, 0.f, 0.f, 0.f, 0.f};
            acc[i][j] = z;
        }

    const int T = K / BK;

#if HAVE_ASYNC_LDS
    // ---- async global->LDS pipeline (no VGPR staging) --------------------
    {
        async_copy_b128(pa0, sa0); async_copy_b128(pa1, sa1);
        async_copy_b128(pb0, sb0); async_copy_b128(pb1, sb1);
    }
    for (int t = 0; t < T; ++t) {
        const int p = t & 1;

        wait_async0();       // own copies into buffer p have landed
        __syncthreads();     // everyone's have; prior reads of p^1 done

        if (t + 1 < T) {     // fill the other buffer during this compute
            const int ko  = (t + 1) * BK;
            const int off = (p ^ 1) * bufStride;
            async_copy_b128(pa0 + ko, sa0 + off);
            async_copy_b128(pa1 + ko, sa1 + off);
            async_copy_b128(pb0 + ko, sb0 + off);
            async_copy_b128(pb1 + ko, sb1 + off);
        }
        if (t + 2 < T) {     // pull tile t+2 toward GL2
            __builtin_prefetch(pa0 + (t + 2) * BK, 0, 1);
            __builtin_prefetch(pb0 + (t + 2) * BK, 0, 1);
        }

        const _Float16* ap = As[p];
        const _Float16* bp = Bs[p];
        union { v16h v; h8 h[2]; } af[2], bf[4];
        #pragma unroll
        for (int i = 0; i < 2; ++i) {
            const int m = wm * 32 + i * 16 + mrow;
            af[i].h[0] = *(const h8*)&ap[m * LDT + cA];
            af[i].h[1] = *(const h8*)&ap[m * LDT + 16 + cA];
        }
        #pragma unroll
        for (int j = 0; j < 4; ++j) {
            const int n = wn * 64 + j * 16 + mrow;
            bf[j].h[0] = *(const h8*)&bp[n * LDT + cA];
            bf[j].h[1] = *(const h8*)&bp[n * LDT + 16 + cA];
        }
        #pragma unroll
        for (int i = 0; i < 2; ++i)
            #pragma unroll
            for (int j = 0; j < 4; ++j)
                acc[i][j] = __builtin_amdgcn_wmma_f32_16x16x32_f16(
                    false, af[i].v, false, bf[j].v,
                    (short)0, acc[i][j], false, false);
    }
#else
    // ---- fallback: register-staged double buffering ----------------------
    h8 ra0, ra1, rb0, rb1;
    ra0 = *(const h8*)(pa0); ra1 = *(const h8*)(pa1);
    rb0 = *(const h8*)(pb0); rb1 = *(const h8*)(pb1);
    *(h8*)sa0 = ra0; *(h8*)sa1 = ra1;
    *(h8*)sb0 = rb0; *(h8*)sb1 = rb1;

    for (int t = 0; t < T; ++t) {
        const int p = t & 1;
        if (t + 1 < T) {
            const int ko = (t + 1) * BK;
            ra0 = *(const h8*)(pa0 + ko); ra1 = *(const h8*)(pa1 + ko);
            rb0 = *(const h8*)(pb0 + ko); rb1 = *(const h8*)(pb1 + ko);
        }
        if (t + 2 < T) {
            __builtin_prefetch(pa0 + (t + 2) * BK, 0, 1);
            __builtin_prefetch(pb0 + (t + 2) * BK, 0, 1);
        }
        __syncthreads();

        const _Float16* ap = As[p];
        const _Float16* bp = Bs[p];
        union { v16h v; h8 h[2]; } af[2], bf[4];
        #pragma unroll
        for (int i = 0; i < 2; ++i) {
            const int m = wm * 32 + i * 16 + mrow;
            af[i].h[0] = *(const h8*)&ap[m * LDT + cA];
            af[i].h[1] = *(const h8*)&ap[m * LDT + 16 + cA];
        }
        #pragma unroll
        for (int j = 0; j < 4; ++j) {
            const int n = wn * 64 + j * 16 + mrow;
            bf[j].h[0] = *(const h8*)&bp[n * LDT + cA];
            bf[j].h[1] = *(const h8*)&bp[n * LDT + 16 + cA];
        }
        #pragma unroll
        for (int i = 0; i < 2; ++i)
            #pragma unroll
            for (int j = 0; j < 4; ++j)
                acc[i][j] = __builtin_amdgcn_wmma_f32_16x16x32_f16(
                    false, af[i].v, false, bf[j].v,
                    (short)0, acc[i][j], false, false);

        if (t + 1 < T) {
            const int off = (p ^ 1) * bufStride;
            *(h8*)(sa0 + off) = ra0; *(h8*)(sa1 + off) = ra1;
            *(h8*)(sb0 + off) = rb0; *(h8*)(sb1 + off) = rb1;
        }
    }
#endif

    // ---- epilogue --------------------------------------------------------
    #pragma unroll
    for (int i = 0; i < 2; ++i) {
        #pragma unroll
        for (int j = 0; j < 4; ++j) {
            const int gr0 = bm + wm * 32 + i * 16 + (half << 3);
            const int gc  = bn + wn * 64 + j * 16 + mrow;
            const float bvl = bias[gc];
            #pragma unroll
            for (int r = 0; r < 8; ++r) {
                const size_t idx = (size_t)(gr0 + r) * N + gc;
                const float a = acc[i][j][r] + bvl;
                if (EPI == 1) {
                    // SiLU via v_rcp_f32 instead of IEEE divide
                    const float s = a * __builtin_amdgcn_rcpf(1.0f + __expf(-a));
                    out16[idx] = (_Float16)s;
                } else {
                    out32[idx] = residual[idx] + 0.5f * a;
                }
            }
        }
    }
}

// ---------------------------------------------------------------------------
// launch
// ---------------------------------------------------------------------------
extern "C" void kernel_launch(void* const* d_in, const int* in_sizes, int n_in,
                              void* d_out, int out_size, void* d_ws, size_t ws_size,
                              hipStream_t stream)
{
    (void)in_sizes; (void)n_in; (void)out_size; (void)ws_size;

    const int Mtot = 8 * 2048;   // B*S
    const int H    = 1024;
    const int F    = 4096;

    const float* x        = (const float*)d_in[0];
    const float* ln_gamma = (const float*)d_in[1];
    const float* ln_beta  = (const float*)d_in[2];
    const int*   fc1_q    = (const int*)d_in[3];
    const float* fc1_s    = (const float*)d_in[4];
    const float* fc1_b    = (const float*)d_in[5];
    const int*   fc2_q    = (const int*)d_in[6];
    const float* fc2_s    = (const float*)d_in[7];
    const float* fc2_b    = (const float*)d_in[8];
    float*       out      = (float*)d_out;

    // workspace carve-up (bytes):
    //   h16  : M*H*2 = 32 MB      w1t : F*H*2 = 8 MB
    //   w2t  : H*F*2 =  8 MB      y16 : M*F*2 = 128 MB
    char* ws = (char*)d_ws;
    _Float16* h16 = (_Float16*)(ws);
    _Float16* w1t = (_Float16*)(ws + ((size_t)32 << 20));   // [F, H]
    _Float16* w2t = (_Float16*)(ws + ((size_t)40 << 20));   // [H, F]
    _Float16* y16 = (_Float16*)(ws + ((size_t)48 << 20));   // [M, F]

    // 1) LayerNorm -> f16
    ffn_ln_kernel<<<Mtot, 256, 0, stream>>>(x, ln_gamma, ln_beta, h16);

    // 2) dequantize + transpose weights -> f16  (wt[N,K])
    {
        dim3 g1(F / TDN, H / TDK);   // (64, 16)
        ffn_dequant_t_kernel<<<g1, 256, 0, stream>>>(fc1_q, fc1_s, w1t, H, F);
        dim3 g2(H / TDN, F / TDK);   // (16, 64)
        ffn_dequant_t_kernel<<<g2, 256, 0, stream>>>(fc2_q, fc2_s, w2t, F, H);
    }

    // 3) FC1 + bias + SiLU -> y16   [M,F]
    {
        dim3 grid(F / BN, Mtot / BM);            // (32, 128)
        ffn_gemm_kernel<1><<<grid, 256, 0, stream>>>(
            h16, w1t, fc1_b, nullptr, y16, nullptr, Mtot, F, H);
    }

    // 4) FC2 + bias + residual + 0.5 -> fp32 out   [M,H]
    {
        dim3 grid(H / BN, Mtot / BM);            // (8, 128)
        ffn_gemm_kernel<2><<<grid, 256, 0, stream>>>(
            y16, w2t, fc2_b, x, nullptr, out, Mtot, H, F);
    }
}
